// SimpleRNN_81698867904724
// MI455X (gfx1250) — matrix operand order repaired and checked by reading, not compiled, not run on hardware
//
#include <hip/hip_runtime.h>
#include <hip/hip_bf16.h>

// ---------------------------------------------------------------------------
// SimpleRNN forward for MI455X (gfx1250, wave32, WMMA + TDM).
//   xproj = x @ W_ih^T + b_ih + b_hh            (big parallel bf16 WMMA GEMM)
//   h_t   = tanh(xproj_t + h_{t-1} @ W_hh^T)    (persistent kernel, grid barrier)
//   y     = (h_T @ W_fc^T + b_fc) @ W_clf^T + b_clf
// GEMMs: bf16 inputs, f32 accumulation via v_wmma_f32_16x16x32_bf16.
// A-slabs are staged into LDS by the Tensor Data Mover (tensor_load_to_lds),
// weights are pre-swizzled into the exact B-fragment lane layout.
// ---------------------------------------------------------------------------

typedef __bf16 bf16;
typedef __attribute__((ext_vector_type(16))) __bf16 v16bf;
typedef __attribute__((ext_vector_type(8)))  float  v8f;
typedef __attribute__((ext_vector_type(4)))  unsigned int v4u;
typedef __attribute__((ext_vector_type(8)))  int v8i;
typedef __attribute__((ext_vector_type(4)))  int v4i;

#define B_DIM 64
#define T_DIM 512
#define I_DIM 1024
#define H_DIM 1024
#define O_DIM 256

// ---------------------------------------------------------------------------
// TDM: DMA `rows` x `row_elems` f32 tile (row stride ld_elems) from global
// into LDS at byte offset lds_off.  D# built per CDNA5 ISA
// §8.3/8.4: group0 = {count, lds_addr, global_addr[56:0], type=2};
// group1 = {data_size=4B, tensor_dim0/1 = tile dims, tile_dim0/1, dim0_stride}.
// Tensor dims == tile dims (tile is fully in-bounds by construction).
// This toolchain exposes the 6-arg builtin (extra descriptor group unused here).
// ---------------------------------------------------------------------------
__device__ __forceinline__ void tdm_load_rows_f32(unsigned lds_off,
                                                  const float* __restrict__ gsrc,
                                                  unsigned rows, unsigned row_elems,
                                                  unsigned ld_elems) {
  unsigned long long ga = (unsigned long long)(size_t)(const void*)gsrc;
  v4u g0;
  g0[0] = 1u;                                            // count=1 (valid), user mode
  g0[1] = lds_off;                                       // LDS byte address
  g0[2] = (unsigned)(ga & 0xffffffffull);                // global_addr[31:0]
  g0[3] = (unsigned)((ga >> 32) & 0x01ffffffull)         // global_addr[56:32]
        | (2u << 30);                                    // type=2 ("image")
  v8i g1;
  g1[0] = (int)(2u << 16);                               // wg_mask=0, data_size=4B
  g1[1] = (int)((row_elems & 0xffffu) << 16);            // tensor_dim0[15:0]
  g1[2] = (int)(((row_elems >> 16) & 0xffffu)            // tensor_dim0[31:16]
        | ((rows & 0xffffu) << 16));                     // tensor_dim1[15:0]
  g1[3] = (int)(((rows >> 16) & 0xffffu)                 // tensor_dim1[31:16]
        | ((row_elems & 0xffffu) << 16));                // tile_dim0
  g1[4] = (int)(rows & 0xffffu);                         // tile_dim1, tile_dim2=0
  g1[5] = (int)ld_elems;                                 // tensor_dim0_stride[31:0]
  g1[6] = 0;                                             // stride hi / dim1_stride lo
  g1[7] = 0;
  v4i g2 = {0, 0, 0, 0};
  v4i g3 = {0, 0, 0, 0};
  v8i g4 = {0, 0, 0, 0, 0, 0, 0, 0};                     // unused extra group
  __builtin_amdgcn_tensor_load_to_lds(g0, g1, g2, g3, g4, 0);
}

// ---------------------------------------------------------------------------
// Fragment coordinate helpers (CDNA5 ISA §7.12.2, wave32):
//  A 16x32 bf16 : lane l holds row m = l&15;  K = e + 16*(l>>4) (this file's order)
//  B 32x16 bf16 : lane l holds col n = l&15;  K = e + 16*(l>>4) (this file's order)
//  C/D 16x16 f32: vgpr r, lane l -> row = r + 8*(l>>4), col = l&15
// ---------------------------------------------------------------------------

// A-fragment from f32 row-major global memory (one contiguous 16-float run per lane).
__device__ __forceinline__ v16bf load_a_frag_f32(const float* __restrict__ X,
                                                 int ld, int m, int k0, int lane) {
  const int hi = lane >> 4;
  const float* row = X + (size_t)m * ld + k0;
  v16bf a;
#pragma unroll
  for (int e = 0; e < 16; ++e) {
    int k = e + 16 * hi;
    a[e] = (bf16)row[k];
  }
  return a;
}

// A-fragment from an f32 row-major LDS slab (rows x ld), converting to bf16.
__device__ __forceinline__ v16bf load_a_frag_lds(const float* lds_base,
                                                 int ld, int k0, int lane) {
  const int hi = lane >> 4;
  const float* row = lds_base + (size_t)(lane & 15) * ld + k0;
  v16bf a;
#pragma unroll
  for (int e = 0; e < 16; ++e) {
    int k = e + 16 * hi;
    a[e] = (bf16)row[k];
  }
  return a;
}

// ---------------------------------------------------------------------------
// Pack an f32 weight matrix W[Hout][K] (row-major) into bf16 B-fragments:
//   P[((nt*KT + kt)*32 + lane)*16 + e] = bf16(W[nt*16 + (lane&15)][kt*32 + e + 16*(lane>>4)])
// ---------------------------------------------------------------------------
__global__ void __launch_bounds__(256)
pack_w_kernel(const float* __restrict__ W, bf16* __restrict__ P, int Hout, int K) {
  const int KT = K >> 5;
  const int NT = Hout >> 4;
  const int wave = (int)((blockIdx.x * blockDim.x + threadIdx.x) >> 5);
  const int lane = (int)(threadIdx.x & 31);
  if (wave >= NT * KT) return;
  const int nt = wave / KT, kt = wave % KT;
  const int n = nt * 16 + (lane & 15);
  const int kbase = kt * 32 + ((lane >> 4) << 4);
  const float* src = W + (size_t)n * K + kbase;
  bf16* dst = P + ((size_t)(nt * KT + kt) * 32 + lane) * 16;
#pragma unroll
  for (int e = 0; e < 16; ++e) dst[e] = (bf16)src[e];
}

// ---------------------------------------------------------------------------
// xproj[t][b][h] = x[b][t][:] . W_ih[h][:] + (b_ih[h] + b_hh[h])
// Block = 8 waves sharing one 16-row M-tile; TDM stages the 16x1024 f32
// A-slab (64KB) into LDS once; each wave then runs 32 chained WMMAs.
// ---------------------------------------------------------------------------
__global__ void __launch_bounds__(256)
xproj_kernel(const float* __restrict__ x, const bf16* __restrict__ WihP,
             const float* __restrict__ b_ih, const float* __restrict__ b_hh,
             float* __restrict__ xproj) {
  __shared__ float alds[16 * I_DIM];      // 64KB A-slab
  const int KT = I_DIM / 32;              // 32
  const int NT = H_DIM / 16;              // 64
  const int wave = (int)((blockIdx.x * blockDim.x + threadIdx.x) >> 5);
  const int lane = (int)(threadIdx.x & 31);
  const int nt = wave & (NT - 1);
  const int mt = wave >> 6;               // constant within a block (8 | 64)
  const int m0 = mt * 16;
  const int n  = nt * 16 + (lane & 15);

  // Wave 0: one TDM op for the whole block's A-slab.
  if (threadIdx.x < 32) {
    tdm_load_rows_f32((unsigned)(size_t)(void*)alds,
                      x + (size_t)m0 * I_DIM, 16u, (unsigned)I_DIM, (unsigned)I_DIM);
    __builtin_amdgcn_s_wait_tensorcnt(0);
  }
  __syncthreads();

  const bf16* bp = WihP + ((size_t)nt * KT * 32 + lane) * 16;
  v8f c = {};
  for (int kt = 0; kt < KT; ++kt) {
    if (kt + 1 < KT)  // prefetch next packed-B slab (global_prefetch_b8)
      __builtin_prefetch((const void*)(bp + (size_t)(kt + 1) * (32 * 16)), 0, 1);
    v16bf a = load_a_frag_lds(alds, I_DIM, kt * 32, lane);
    v16bf b = *(const v16bf*)(bp + (size_t)kt * (32 * 16));
    c = __builtin_amdgcn_wmma_f32_16x16x32_bf16(false, a, false, b, (short)0, c,
                                                false, false);
  }

  const float bias = b_ih[n] + b_hh[n];
  const int rbase = m0 + 8 * (lane >> 4);
#pragma unroll
  for (int r = 0; r < 8; ++r) {
    int row = rbase + r;               // row in [0, B*T) of x viewed as [B*T, I]
    int bb = row >> 9;                 // / T_DIM
    int tt = row & (T_DIM - 1);
    xproj[((size_t)tt * B_DIM + bb) * H_DIM + n] = c[r] + bias;
  }
}

// ---------------------------------------------------------------------------
// Software grid barrier (cumulative-count, no reset races).
// bar[0]: arrival count, bar[1]: generation. Both zeroed by init_kernel.
// ---------------------------------------------------------------------------
__device__ __forceinline__ void grid_barrier(unsigned* __restrict__ bar,
                                             unsigned nblocks, unsigned& local_gen) {
  __threadfence();
  __syncthreads();
  if (threadIdx.x == 0) {
    unsigned arrived = atomicAdd(&bar[0], 1u) + 1u;
    if (arrived == nblocks * (local_gen + 1u)) {
      atomicAdd(&bar[1], 1u);  // release this generation
    }
    while (atomicAdd(&bar[1], 0u) <= local_gen) {
      __builtin_amdgcn_s_sleep(1);
    }
  }
  local_gen++;
  __syncthreads();
  __threadfence();
}

// ---------------------------------------------------------------------------
// Persistent recurrence: 512 steps of h' = tanh(xproj_t + h @ W_hh^T).
// 256 waves (4 M-tiles x 64 N-tiles), one 16x16 tile per wave.  Each step:
// TDM stages the block's 16 h-rows (64KB f32) into LDS, 32 chained WMMAs,
// fused bias/tanh epilogue, then a device-wide barrier.
// ---------------------------------------------------------------------------
__global__ void __launch_bounds__(256)
scan_kernel(const bf16* __restrict__ WhhP, const float* __restrict__ xproj,
            float* __restrict__ h0, float* __restrict__ h1,
            unsigned* __restrict__ bar) {
  __shared__ float alds[16 * H_DIM];      // 64KB h-slab
  const int KT = H_DIM / 32;              // 32
  const int NT = H_DIM / 16;              // 64
  const int wave = (int)((blockIdx.x * blockDim.x + threadIdx.x) >> 5);
  const int lane = (int)(threadIdx.x & 31);
  const int nt = wave & (NT - 1);
  const int mt = wave >> 6;               // 0..3, constant within a block
  const int m0 = mt * 16;
  const int n  = nt * 16 + (lane & 15);
  const int rbase = m0 + 8 * (lane >> 4);

  const bf16* bp = WhhP + ((size_t)nt * KT * 32 + lane) * 16;
  float* hbuf[2] = {h0, h1};
  unsigned local_gen = 0;

  for (int t = 0; t < T_DIM; ++t) {
    const float* hp = hbuf[t & 1];
    float* hn = hbuf[(t + 1) & 1];

    // Stage this block's 16 rows of h_prev into LDS via the TDM.
    if (threadIdx.x < 32) {
      tdm_load_rows_f32((unsigned)(size_t)(void*)alds,
                        hp + (size_t)m0 * H_DIM, 16u, (unsigned)H_DIM, (unsigned)H_DIM);
      __builtin_amdgcn_s_wait_tensorcnt(0);
    }
    __syncthreads();

    v8f c = {};
    for (int kt = 0; kt < KT; ++kt) {
      v16bf a = load_a_frag_lds(alds, H_DIM, kt * 32, lane);
      v16bf b = *(const v16bf*)(bp + (size_t)kt * (32 * 16));
      c = __builtin_amdgcn_wmma_f32_16x16x32_bf16(false, a, false, b, (short)0, c,
                                                  false, false);
    }

    const float* xp = xproj + (size_t)t * B_DIM * H_DIM;
#pragma unroll
    for (int r = 0; r < 8; ++r) {
      int row = rbase + r;
      hn[(size_t)row * H_DIM + n] = tanhf(c[r] + xp[(size_t)row * H_DIM + n]);
    }

    grid_barrier(bar, gridDim.x, local_gen);  // also orders LDS reuse next step
  }
}

// ---------------------------------------------------------------------------
// Generic C[M,N] = A[M,K] @ Wpacked^T + bias, one wave per 16x16 tile.
// Used for fc (N=1024) and clf (N=256, writes d_out).
// ---------------------------------------------------------------------------
__global__ void __launch_bounds__(256)
gemm_bias_kernel(const float* __restrict__ A, const bf16* __restrict__ BP,
                 const float* __restrict__ bias, float* __restrict__ C,
                 int M, int N, int K) {
  const int KT = K >> 5;
  const int NT = N >> 4;
  const int wave = (int)((blockIdx.x * blockDim.x + threadIdx.x) >> 5);
  const int lane = (int)(threadIdx.x & 31);
  if (wave >= (M >> 4) * NT) return;   // whole-wave branch: EXEC stays all-1s
  const int nt = wave % NT, mt = wave / NT;
  const int m0 = mt * 16;
  const int mA = m0 + (lane & 15);
  const int n  = nt * 16 + (lane & 15);

  const bf16* bp = BP + ((size_t)nt * KT * 32 + lane) * 16;
  v8f c = {};
  for (int kt = 0; kt < KT; ++kt) {
    v16bf a = load_a_frag_f32(A, K, mA, kt * 32, lane);
    v16bf b = *(const v16bf*)(bp + (size_t)kt * (32 * 16));
    c = __builtin_amdgcn_wmma_f32_16x16x32_bf16(false, a, false, b, (short)0, c,
                                                false, false);
  }
  const float bb = bias[n];
  const int rbase = m0 + 8 * (lane >> 4);
#pragma unroll
  for (int r = 0; r < 8; ++r) {
    C[(size_t)(rbase + r) * N + n] = c[r] + bb;
  }
}

// Zero h ping-pong buffers and the grid-barrier words (deterministic per call).
__global__ void __launch_bounds__(256)
init_kernel(float* __restrict__ h0, float* __restrict__ h1, unsigned* __restrict__ bar) {
  int i = (int)(blockIdx.x * blockDim.x + threadIdx.x);
  if (i < B_DIM * H_DIM) { h0[i] = 0.0f; h1[i] = 0.0f; }
  if (i < 2) bar[i] = 0u;
}

// ---------------------------------------------------------------------------
extern "C" void kernel_launch(void* const* d_in, const int* in_sizes, int n_in,
                              void* d_out, int out_size, void* d_ws, size_t ws_size,
                              hipStream_t stream) {
  const float* x     = (const float*)d_in[0];
  const float* W_ih  = (const float*)d_in[1];
  const float* b_ih  = (const float*)d_in[2];
  const float* W_hh  = (const float*)d_in[3];
  const float* b_hh  = (const float*)d_in[4];
  const float* W_fc  = (const float*)d_in[5];
  const float* b_fc  = (const float*)d_in[6];
  const float* W_clf = (const float*)d_in[7];
  const float* b_clf = (const float*)d_in[8];
  float* out = (float*)d_out;

  // Workspace carve-out (256B aligned chunks).
  char* ws = (char*)d_ws;
  size_t off = 0;
  auto carve = [&](size_t bytes) -> char* {
    char* p = ws + off;
    off += (bytes + 255) & ~(size_t)255;
    return p;
  };
  bf16*  WihP  = (bf16*)carve((size_t)H_DIM * I_DIM * sizeof(bf16));
  bf16*  WhhP  = (bf16*)carve((size_t)H_DIM * H_DIM * sizeof(bf16));
  bf16*  WfcP  = (bf16*)carve((size_t)H_DIM * H_DIM * sizeof(bf16));
  bf16*  WclfP = (bf16*)carve((size_t)O_DIM * H_DIM * sizeof(bf16));
  float* xproj = (float*)carve((size_t)T_DIM * B_DIM * H_DIM * sizeof(float));
  float* h0    = (float*)carve((size_t)B_DIM * H_DIM * sizeof(float));
  float* h1    = (float*)carve((size_t)B_DIM * H_DIM * sizeof(float));
  float* y1    = (float*)carve((size_t)B_DIM * H_DIM * sizeof(float));
  unsigned* bar = (unsigned*)carve(256);

  // 1) Pack all weights into bf16 B-fragment layout (waves = NT*KT).
  pack_w_kernel<<<(64 * 32 + 7) / 8, 256, 0, stream>>>(W_ih,  WihP,  H_DIM, I_DIM);
  pack_w_kernel<<<(64 * 32 + 7) / 8, 256, 0, stream>>>(W_hh,  WhhP,  H_DIM, H_DIM);
  pack_w_kernel<<<(64 * 32 + 7) / 8, 256, 0, stream>>>(W_fc,  WfcP,  H_DIM, H_DIM);
  pack_w_kernel<<<(16 * 32 + 7) / 8, 256, 0, stream>>>(W_clf, WclfP, O_DIM, H_DIM);

  // 2) Zero h buffers + barrier state.
  init_kernel<<<(B_DIM * H_DIM + 255) / 256, 256, 0, stream>>>(h0, h1, bar);

  // 3) Big GEMM: xproj for all timesteps.  (B*T/16) * (H/16) = 131072 waves.
  xproj_kernel<<<131072 / 8, 256, 0, stream>>>(x, WihP, b_ih, b_hh, xproj);

  // 4) Persistent recurrence: 256 waves = 32 blocks of 8 waves, 512 steps.
  scan_kernel<<<32, 256, 0, stream>>>(WhhP, xproj, h0, h1, bar);

  // 5) Tail GEMMs: h_last lives in h0 after an even number of steps.
  gemm_bias_kernel<<<32, 256, 0, stream>>>(h0, WfcP, b_fc, y1, B_DIM, H_DIM, H_DIM);
  gemm_bias_kernel<<<8, 256, 0, stream>>>(y1, WclfP, b_clf, out, B_DIM, O_DIM, H_DIM);
}
